// CVCM_43241730736365
// MI455X (gfx1250) — matrix operand order repaired and checked by reading, not compiled, not run on hardware
//
#include <hip/hip_runtime.h>
#include <hip/hip_bf16.h>
#include <math.h>

typedef __attribute__((ext_vector_type(16))) _Float16 v16h;
typedef __attribute__((ext_vector_type(8)))  _Float16 v8h;
typedef __attribute__((ext_vector_type(8)))  float    v8f;

#define BSZ 32
#define LP  96
#define DM  12
#define ED  768
#define NST 8
#define DC  3
#define NL  10
#define EMB 256
#define PP  50
#define ROWS (BSZ*LP)          // 3072
#define XZW (2*ED)             // 1536

__device__ __forceinline__ float sigf(float x) { return 1.f / (1.f + __expf(-x)); }
__device__ __forceinline__ float siluf(float x) { return x * sigf(x); }

__device__ __forceinline__ v16h ld16h(const _Float16* p0, const _Float16* p1) {
  v8h lo = *(const v8h*)p0;
  v8h hi = *(const v8h*)p1;
  return __builtin_shufflevector(lo, hi, 0,1,2,3,4,5,6,7,8,9,10,11,12,13,14,15);
}

// ---------------- K0: complex front conv + pos -> h (ROWS, DM) ----------------
__global__ void k_front(const float* __restrict__ x,
                        const float* __restrict__ rew, const float* __restrict__ reb,
                        const float* __restrict__ imw, const float* __restrict__ imb,
                        const float* __restrict__ pos, float* __restrict__ h) {
  int t = blockIdx.x * blockDim.x + threadIdx.x;       // ROWS*6 = 18432
  if (t >= ROWS * (DM/2)) return;
  int o = t % (DM/2);
  int bl = t / (DM/2);
  int b = bl / LP, l = bl % LP;
  const float* xr = x + (size_t)(b*2 + 0)*4800 + l*PP;
  const float* xi = x + (size_t)(b*2 + 1)*4800 + l*PP;
  float srr = 0.f, sir = 0.f, sri = 0.f, sii = 0.f;
  #pragma unroll 5
  for (int k = 0; k < PP; ++k) {
    float xrv = xr[k], xiv = xi[k];
    float wr = rew[o*PP + k], wi = imw[o*PP + k];
    srr += xrv * wr; sir += xiv * wr;
    sri += xrv * wi; sii += xiv * wi;
  }
  float re_r = srr + reb[o], im_r = sir + reb[o];
  float re_i = sri + imb[o], im_i = sii + imb[o];
  h[bl*DM + o]        = (re_r - im_i) + pos[l*DM + o];
  h[bl*DM + DM/2 + o] = (im_r + re_i) + pos[l*DM + DM/2 + o];
}

// ---------------- K1: batchnorm over (ROWS, DM) + SiLU, single block ----------------
__global__ void k_bn_silu(float* __restrict__ h,
                          const float* __restrict__ gamma, const float* __restrict__ beta) {
  __shared__ float ssum[256*DM];
  __shared__ float ssq [256*DM];
  __shared__ float sscale[DM], sshift[DM];
  int tid = threadIdx.x;
  float ls[DM], lq[DM];
  #pragma unroll
  for (int d = 0; d < DM; ++d) { ls[d] = 0.f; lq[d] = 0.f; }
  for (int row = tid; row < ROWS; row += 256) {
    #pragma unroll
    for (int d = 0; d < DM; ++d) {
      float v = h[row*DM + d];
      ls[d] += v; lq[d] += v * v;
    }
  }
  #pragma unroll
  for (int d = 0; d < DM; ++d) { ssum[tid*DM + d] = ls[d]; ssq[tid*DM + d] = lq[d]; }
  __syncthreads();
  for (int s = 128; s > 0; s >>= 1) {
    if (tid < s) {
      #pragma unroll
      for (int d = 0; d < DM; ++d) {
        ssum[tid*DM + d] += ssum[(tid+s)*DM + d];
        ssq [tid*DM + d] += ssq [(tid+s)*DM + d];
      }
    }
    __syncthreads();
  }
  if (tid < DM) {
    float mu  = ssum[tid] / (float)ROWS;
    float var = ssq[tid] / (float)ROWS - mu * mu;
    float sc  = rsqrtf(var + 1e-6f) * gamma[tid];
    sscale[tid] = sc;
    sshift[tid] = beta[tid] - mu * sc;
  }
  __syncthreads();
  for (int idx = tid; idx < ROWS*DM; idx += 256) {
    int d = idx % DM;
    float v = h[idx] * sscale[d] + sshift[d];
    h[idx] = siluf(v);
  }
}

// ---------------- pad + f32->f16 convert: dst(dr,dc) from src(sr,sc) ----------------
__global__ void k_cvt_pad(const float* __restrict__ src, _Float16* __restrict__ dst,
                          int sr, int sc, int dr, int dc) {
  int t = blockIdx.x * blockDim.x + threadIdx.x;
  if (t >= dr*dc) return;
  int r = t / dc, c = t % dc;
  float v = (r < sr && c < sc) ? src[r*sc + c] : 0.f;
  dst[t] = (_Float16)v;
}

// ---------------- K2: RMSNorm -> u16 (ROWS, 32) zero-padded f16 ----------------
__global__ void k_rms(const float* __restrict__ h, const float* __restrict__ rw,
                      _Float16* __restrict__ u16) {
  int row = blockIdx.x * blockDim.x + threadIdx.x;
  if (row >= ROWS) return;
  float v[DM]; float ss = 0.f;
  #pragma unroll
  for (int d = 0; d < DM; ++d) { v[d] = h[row*DM + d]; ss += v[d]*v[d]; }
  float sc = rsqrtf(ss / (float)DM + 1e-5f);
  v8h outv[4];
  #pragma unroll
  for (int c = 0; c < 4; ++c) {
    #pragma unroll
    for (int j = 0; j < 8; ++j) {
      int d = c*8 + j;                                   // compile-time per iteration
      outv[c][j] = (d < DM) ? (_Float16)(v[d] * sc * rw[d]) : (_Float16)0.f;
    }
  }
  v8h* dst = (v8h*)(u16 + (size_t)row*32);
  #pragma unroll
  for (int c = 0; c < 4; ++c) dst[c] = outv[c];
}

// ---------------- K3: in_proj WMMA: xz(3072,1536) = u16(3072,32p) @ W16(1536,32p)^T ----
__global__ void k_inproj_wmma(const _Float16* __restrict__ u16,
                              const _Float16* __restrict__ w16,
                              float* __restrict__ xz) {
  int wave = (blockIdx.x * blockDim.x + threadIdx.x) >> 5;   // 0..4607
  int lane = threadIdx.x & 31;
  int mt = wave / 24;          // 192 M-tiles
  int ng = wave % 24;          // 24 groups of 4 N-tiles
  int r  = lane & 15;
  int hh = lane >> 4;
  int m0 = mt * 16;

  const _Float16* arow = u16 + (size_t)(m0 + r)*32 + hh*16;
  v16h a = ld16h(arow, arow + 8);

  #pragma unroll
  for (int t = 0; t < 4; ++t) {
    int n0 = (ng*4 + t) * 16;
    const _Float16* brow = w16 + (size_t)(n0 + r)*32 + hh*16;
    v16h b = ld16h(brow, brow + 8);
    v8f c = {};
    c = __builtin_amdgcn_wmma_f32_16x16x32_f16(false, a, false, b, (short)0, c, false, false);
    #pragma unroll
    for (int v = 0; v < 8; ++v) {
      int m = m0 + v + 8*hh;
      xz[(size_t)m*XZW + n0 + r] = c[v];
    }
  }
}

// ---------------- K4: depthwise causal conv1d + SiLU -> xc (f32) and xc16 (f16) ------
__global__ void k_conv1d(const float* __restrict__ xz, const float* __restrict__ cw,
                         const float* __restrict__ cb, float* __restrict__ xc,
                         _Float16* __restrict__ xc16) {
  int t = blockIdx.x * blockDim.x + threadIdx.x;        // ROWS*ED
  if (t >= ROWS*ED) return;
  int e = t % ED;
  int row = t / ED;
  int l = row % LP, b = row / LP;
  float acc = cb[e];
  #pragma unroll
  for (int k = 0; k < DC; ++k) {
    int ls = l - (DC - 1) + k;
    if (ls >= 0) acc += cw[e*DC + k] * xz[(size_t)(b*LP + ls)*XZW + e];
  }
  float s = siluf(acc);
  xc  [(size_t)row*ED + e] = s;
  xc16[(size_t)row*ED + e] = (_Float16)s;
}

// ---------------- K5: x_proj WMMA: dblp(3072,32p) = xc16 @ W16(32p,768)^T ------------
// wave per M-tile (192 waves); 2 N-tiles per wave reusing A; K loop 24 x 32
__global__ void k_xproj_wmma(const _Float16* __restrict__ xc16,
                             const _Float16* __restrict__ w16,
                             float* __restrict__ dblp) {
  int wave = (blockIdx.x * blockDim.x + threadIdx.x) >> 5;   // 0..191
  int lane = threadIdx.x & 31;
  int r  = lane & 15;
  int hh = lane >> 4;
  int m0 = wave * 16;
  v8f c0 = {}, c1 = {};
  for (int kc = 0; kc < ED/32; ++kc) {
    int kb = kc * 32;
    const _Float16* arow = xc16 + (size_t)(m0 + r)*ED + kb + hh*16;
    v16h a = ld16h(arow, arow + 8);
    const _Float16* b0p = w16 + (size_t)r*ED + kb + hh*16;
    v16h b0 = ld16h(b0p, b0p + 8);
    const _Float16* b1p = w16 + (size_t)(16 + r)*ED + kb + hh*16;
    v16h b1 = ld16h(b1p, b1p + 8);
    c0 = __builtin_amdgcn_wmma_f32_16x16x32_f16(false, a, false, b0, (short)0, c0, false, false);
    c1 = __builtin_amdgcn_wmma_f32_16x16x32_f16(false, a, false, b1, (short)0, c1, false, false);
  }
  #pragma unroll
  for (int v = 0; v < 8; ++v) {
    int m = m0 + v + 8*hh;
    dblp[m*32 + r]      = c0[v];
    dblp[m*32 + 16 + r] = c1[v];
  }
}

// ---------------- K6: selective scan, fused delta/softplus + gate -> y16 -------------
__global__ void k_scan(const float* __restrict__ dblp, const float* __restrict__ xc,
                       const float* __restrict__ xz, const float* __restrict__ dtw,
                       const float* __restrict__ dtb, const float* __restrict__ Alog,
                       const float* __restrict__ Dp, _Float16* __restrict__ y16) {
  int t = blockIdx.x * blockDim.x + threadIdx.x;        // BSZ*ED = 24576
  if (t >= BSZ*ED) return;
  int b = t / ED, e = t % ED;
  float Ae[NST];
  #pragma unroll
  for (int n = 0; n < NST; ++n) Ae[n] = -__expf(Alog[e*NST + n]);
  float w = dtw[e], bias = dtb[e], Dv = Dp[e];
  float st[NST];
  #pragma unroll
  for (int n = 0; n < NST; ++n) st[n] = 0.f;
  for (int l = 0; l < LP; ++l) {
    int row = b*LP + l;
    const float* dr = dblp + row*32;                    // wave-uniform row
    float draw = dr[0] * w + bias;
    float delta = (draw > 20.f) ? draw : log1pf(__expf(draw));
    float xcv = xc[(size_t)row*ED + e];
    float zv  = xz[(size_t)row*XZW + ED + e];
    float du = delta * xcv;
    float acc = 0.f;
    #pragma unroll
    for (int n = 0; n < NST; ++n) {
      float dA = __expf(delta * Ae[n]);
      st[n] = dA * st[n] + du * dr[1 + n];
      acc += st[n] * dr[1 + NST + n];
    }
    float yv = acc + Dv * xcv;
    y16[(size_t)row*ED + e] = (_Float16)(yv * siluf(zv));
  }
}

// ---------------- K7: out_proj WMMA + residual: h += y16(3072,768) @ W16(16p,768)^T --
__global__ void k_outproj_wmma(const _Float16* __restrict__ y16,
                               const _Float16* __restrict__ w16,
                               float* __restrict__ h) {
  int wave = (blockIdx.x * blockDim.x + threadIdx.x) >> 5;   // 0..191
  int lane = threadIdx.x & 31;
  int r  = lane & 15;
  int hh = lane >> 4;
  int m0 = wave * 16;
  v8f c = {};
  for (int kc = 0; kc < ED/32; ++kc) {
    int kb = kc * 32;
    const _Float16* arow = y16 + (size_t)(m0 + r)*ED + kb + hh*16;
    if (kc + 1 < ED/32) __builtin_prefetch(arow + 32, 0, 1);
    v16h a = ld16h(arow, arow + 8);
    const _Float16* brow = w16 + (size_t)r*ED + kb + hh*16;
    v16h b = ld16h(brow, brow + 8);
    c = __builtin_amdgcn_wmma_f32_16x16x32_f16(false, a, false, b, (short)0, c, false, false);
  }
  #pragma unroll
  for (int v = 0; v < 8; ++v) {
    int m = m0 + v + 8*hh;
    if (r < DM) h[m*DM + r] += c[v];
  }
}

// ---------------- K8a: mean over L -> mbuf (BSZ, DM) ----------------
__global__ void k_mean(const float* __restrict__ h, float* __restrict__ mbuf) {
  int t = blockIdx.x * blockDim.x + threadIdx.x;
  if (t >= BSZ*DM) return;
  int b = t / DM, d = t % DM;
  float acc = 0.f;
  for (int l = 0; l < LP; ++l) acc += h[(b*LP + l)*DM + d];
  mbuf[b*DM + d] = acc / (float)LP;
}

// ---------------- K8b: fc + relu -> out (BSZ, EMB) ----------------
__global__ void k_fc(const float* __restrict__ mbuf, const float* __restrict__ fcw,
                     const float* __restrict__ fcb, float* __restrict__ out) {
  int t = blockIdx.x * blockDim.x + threadIdx.x;        // BSZ*EMB
  if (t >= BSZ*EMB) return;
  int b = t / EMB, j = t % EMB;
  float acc = fcb[j];
  #pragma unroll
  for (int d = 0; d < DM; ++d) acc += mbuf[b*DM + d] * fcw[j*DM + d];
  out[b*EMB + j] = fmaxf(acc, 0.f);
}

extern "C" void kernel_launch(void* const* d_in, const int* in_sizes, int n_in,
                              void* d_out, int out_size, void* d_ws, size_t ws_size,
                              hipStream_t stream) {
  const float* x    = (const float*)d_in[0];
  const float* rew  = (const float*)d_in[1];
  const float* reb  = (const float*)d_in[2];
  const float* imw  = (const float*)d_in[3];
  const float* imb  = (const float*)d_in[4];
  const float* pos  = (const float*)d_in[5];
  const float* bng  = (const float*)d_in[6];
  const float* bnb  = (const float*)d_in[7];
  const float* rmsw = (const float*)d_in[8];
  const float* inpw = (const float*)d_in[9];
  const float* c1w  = (const float*)d_in[10];
  const float* c1b  = (const float*)d_in[11];
  const float* xpw  = (const float*)d_in[12];
  const float* dtw  = (const float*)d_in[13];
  const float* dtb  = (const float*)d_in[14];
  const float* alog = (const float*)d_in[15];
  const float* Dp   = (const float*)d_in[16];
  const float* outw = (const float*)d_in[17];
  const float* fcw  = (const float*)d_in[18];
  const float* fcb  = (const float*)d_in[19];
  float* out = (float*)d_out;

  // ---- workspace layout: f32 region, then f16 region ----
  float* ws = (float*)d_ws;
  float* hbuf  = ws;                         // ROWS*DM
  float* xzbuf = hbuf  + ROWS*DM;            // ROWS*XZW
  float* xcbuf = xzbuf + (size_t)ROWS*XZW;   // ROWS*ED
  float* dblp  = xcbuf + (size_t)ROWS*ED;    // ROWS*32
  float* mbuf  = dblp  + (size_t)ROWS*32;    // BSZ*DM
  _Float16* hp = (_Float16*)(mbuf + BSZ*DM);
  _Float16* u16    = hp;                     // ROWS*32
  _Float16* win16  = u16   + (size_t)ROWS*32;    // XZW*32
  _Float16* xc16   = win16 + (size_t)XZW*32;     // ROWS*ED
  _Float16* wxp16  = xc16  + (size_t)ROWS*ED;    // 32*ED
  _Float16* y16    = wxp16 + (size_t)32*ED;      // ROWS*ED
  _Float16* wout16 = y16   + (size_t)ROWS*ED;    // 16*ED

  // front end
  k_front<<<(ROWS*(DM/2) + 255)/256, 256, 0, stream>>>(x, rew, reb, imw, imb, pos, hbuf);
  k_bn_silu<<<1, 256, 0, stream>>>(hbuf, bng, bnb);

  for (int i = 0; i < NL; ++i) {
    const float* rw_i  = rmsw + i*DM;
    const float* ipw_i = inpw + (size_t)i*XZW*DM;
    const float* c1w_i = c1w + (size_t)i*ED*DC;
    const float* c1b_i = c1b + (size_t)i*ED;
    const float* xpw_i = xpw + (size_t)i*17*ED;
    const float* dtw_i = dtw + (size_t)i*ED;
    const float* dtb_i = dtb + (size_t)i*ED;
    const float* al_i  = alog + (size_t)i*ED*NST;
    const float* D_i   = Dp + (size_t)i*ED;
    const float* opw_i = outw + (size_t)i*DM*ED;

    // per-layer weight prep (pad + f16 convert)
    k_cvt_pad<<<(XZW*32 + 255)/256, 256, 0, stream>>>(ipw_i, win16, XZW, DM, XZW, 32);
    k_cvt_pad<<<(32*ED + 255)/256, 256, 0, stream>>>(xpw_i, wxp16, 17, ED, 32, ED);
    k_cvt_pad<<<(16*ED + 255)/256, 256, 0, stream>>>(opw_i, wout16, DM, ED, 16, ED);

    k_rms<<<(ROWS + 255)/256, 256, 0, stream>>>(hbuf, rw_i, u16);
    // 192 Mtiles * 24 Ngroups = 4608 waves; 8 waves/block -> 576 blocks
    k_inproj_wmma<<<576, 256, 0, stream>>>(u16, win16, xzbuf);
    k_conv1d<<<(ROWS*ED + 255)/256, 256, 0, stream>>>(xzbuf, c1w_i, c1b_i, xcbuf, xc16);
    // 192 waves, 8 waves/block -> 24 blocks
    k_xproj_wmma<<<24, 256, 0, stream>>>(xc16, wxp16, dblp);
    k_scan<<<(BSZ*ED + 255)/256, 256, 0, stream>>>(dblp, xcbuf, xzbuf, dtw_i, dtb_i,
                                                   al_i, D_i, y16);
    k_outproj_wmma<<<24, 256, 0, stream>>>(y16, wout16, hbuf);
  }

  k_mean<<<(BSZ*DM + 127)/128, 128, 0, stream>>>(hbuf, mbuf);
  k_fc<<<(BSZ*EMB + 255)/256, 256, 0, stream>>>(mbuf, fcw, fcb, out);
}
